// MEFI_3882650437010
// MI455X (gfx1250) — compile-verified, matrix operand order checked by reading, not run on hardware
//
#include <hip/hip_runtime.h>

// ---------------------------------------------------------------------------
// MEFI (Mamba-style block) for gfx1250: WMMA f16->f32 GEMMs + VALU scan.
// GEMM tiles staged with GLOBAL_LOAD_ASYNC_TO_LDS (double-buffered) when the
// toolchain exposes the gfx1250 async builtins; sync staging otherwise.
// ---------------------------------------------------------------------------

typedef __attribute__((ext_vector_type(16))) _Float16 v16h;
typedef __attribute__((ext_vector_type(8)))  _Float16 v8h_t;
typedef __attribute__((ext_vector_type(8)))  float    v8f;
typedef __attribute__((ext_vector_type(4)))  int      v4i;

// pointer-to-v4i in global (AS1) / LDS (AS3) address spaces, as required by
// __builtin_amdgcn_global_load_async_to_lds_b128 ("vV4i*1V4i*3IiIi")
typedef __attribute__((address_space(1))) v4i* gv4i_p;
typedef __attribute__((address_space(3))) v4i* lv4i_p;

#define GTM 128
#define GTN 128
#define GTK 32
#define LDS_STRIDE (GTK + 8)   // 40 halfs = 80 bytes per row (16B aligned)

#if __has_builtin(__builtin_amdgcn_global_load_async_to_lds_b128) && \
    __has_builtin(__builtin_amdgcn_s_wait_asynccnt)
#define HAVE_ASYNC 1
#else
#define HAVE_ASYNC 0
#endif

enum { EPI_F32 = 0, EPI_F16 = 1, EPI_RELU_F16 = 2 };

// C[M,N] = A[M,K] * B^T  where Bt is N x K row-major. f16 in, f32 accumulate.
template <int EPI>
__global__ __launch_bounds__(256)
void gemm_wmma_f16(const _Float16* __restrict__ A, int lda,
                   const _Float16* __restrict__ Bt, int ldb,
                   void* __restrict__ Cout, int ldc,
                   int M, int N, int K)
{
    __shared__ alignas(16) _Float16 sA[2][GTM * LDS_STRIDE];
    __shared__ alignas(16) _Float16 sB[2][GTN * LDS_STRIDE];

    const int tid  = threadIdx.x;
    const int lane = tid & 31;
    const int wave = tid >> 5;          // 0..7
    const int wM   = (wave & 3) * 32;   // 4 waves along M
    const int wN   = (wave >> 2) * 64;  // 2 waves along N

    const int blockM = blockIdx.y * GTM;
    const int blockN = blockIdx.x * GTN;

    const int mfrag = lane & 15;
    const int kSel  = (lane >> 4) * 8;  // ISA 7.12.2 16-bit A/B layout

    v8f acc[2][4];
#pragma unroll
    for (int i = 0; i < 2; ++i)
#pragma unroll
        for (int j = 0; j < 4; ++j) {
            v8f z = {};
            acc[i][j] = z;
        }

    const int ldRow = tid >> 1;         // 0..127
    const int ldCol = (tid & 1) * 16;   // 0 or 16
    const int nrowB = blockN + ldRow;
    // clamped row for the async path (garbage only reaches C cols >= N,
    // which the epilogue guard discards)
    const int nrowBc = (nrowB < N) ? nrowB : (N - 1);

    const int nK = K / GTK;

#if HAVE_ASYNC
    // issue one tile's async global->LDS copies (4 b128 ops per thread/wave)
    auto issue_tile = [&](int kt, int buf) {
        const _Float16* srcA = A + (size_t)(blockM + ldRow) * lda + kt * GTK + ldCol;
        _Float16*       dstA = &sA[buf][ldRow * LDS_STRIDE + ldCol];
        const _Float16* srcB = Bt + (size_t)nrowBc * ldb + kt * GTK + ldCol;
        _Float16*       dstB = &sB[buf][ldRow * LDS_STRIDE + ldCol];
        __builtin_amdgcn_global_load_async_to_lds_b128(
            (gv4i_p)(const void*)srcA, (lv4i_p)(void*)dstA, 0, 0);
        __builtin_amdgcn_global_load_async_to_lds_b128(
            (gv4i_p)(const void*)srcA, (lv4i_p)(void*)dstA, 16, 0);
        __builtin_amdgcn_global_load_async_to_lds_b128(
            (gv4i_p)(const void*)srcB, (lv4i_p)(void*)dstB, 0, 0);
        __builtin_amdgcn_global_load_async_to_lds_b128(
            (gv4i_p)(const void*)srcB, (lv4i_p)(void*)dstB, 16, 0);
    };
    issue_tile(0, 0);
#endif

    for (int kt = 0; kt < nK; ++kt) {
        const int cur = kt & 1;
#if HAVE_ASYNC
        if (kt + 1 < nK) {
            __syncthreads();   // all waves done reading buf[cur^1] (iter kt-1)
            issue_tile(kt + 1, cur ^ 1);
            __builtin_amdgcn_s_wait_asynccnt(4);   // tile kt landed (this wave)
        } else {
            __builtin_amdgcn_s_wait_asynccnt(0);
        }
        __syncthreads();       // tile kt visible to all waves
#else
        const int k0 = kt * GTK;
        {
            const _Float16* src = A + (size_t)(blockM + ldRow) * lda + k0 + ldCol;
            v8h_t a0 = *(const v8h_t*)src;
            v8h_t a1 = *(const v8h_t*)(src + 8);
            *(v8h_t*)&sA[cur][ldRow * LDS_STRIDE + ldCol]     = a0;
            *(v8h_t*)&sA[cur][ldRow * LDS_STRIDE + ldCol + 8] = a1;
        }
        {
            v8h_t b0 = {}; v8h_t b1 = {};
            if (nrowB < N) {
                const _Float16* src = Bt + (size_t)nrowB * ldb + k0 + ldCol;
                b0 = *(const v8h_t*)src;
                b1 = *(const v8h_t*)(src + 8);
            }
            *(v8h_t*)&sB[cur][ldRow * LDS_STRIDE + ldCol]     = b0;
            *(v8h_t*)&sB[cur][ldRow * LDS_STRIDE + ldCol + 8] = b1;
        }
        __syncthreads();
#endif

        // ---- load fragments (two 16B LDS reads each; K interleave per ISA) ----
        v16h aF[2], bF[4];
#pragma unroll
        for (int i = 0; i < 2; ++i) {
            const _Float16* p = &sA[cur][(wM + i * 16 + mfrag) * LDS_STRIDE + kSel];
            union { v16h v; v8h_t h[2]; } u;
            u.h[0] = *(const v8h_t*)p;
            u.h[1] = *(const v8h_t*)(p + 16);
            aF[i] = u.v;
        }
#pragma unroll
        for (int j = 0; j < 4; ++j) {
            const _Float16* p = &sB[cur][(wN + j * 16 + mfrag) * LDS_STRIDE + kSel];
            union { v16h v; v8h_t h[2]; } u;
            u.h[0] = *(const v8h_t*)p;
            u.h[1] = *(const v8h_t*)(p + 16);
            bF[j] = u.v;
        }

#pragma unroll
        for (int i = 0; i < 2; ++i)
#pragma unroll
            for (int j = 0; j < 4; ++j)
                acc[i][j] = __builtin_amdgcn_wmma_f32_16x16x32_f16(
                    false, aF[i], false, bF[j], (short)0, acc[i][j], false, false);

#if !HAVE_ASYNC
        __syncthreads();
#endif
    }

    // ---- epilogue: C layout lane=(n), reg r=(m) per ISA 7.12.2 ----
    const int cn0    = blockN + wN + (lane & 15);
    const int cmBase = blockM + wM + ((lane >> 4) * 8);
#pragma unroll
    for (int i = 0; i < 2; ++i) {
#pragma unroll
        for (int j = 0; j < 4; ++j) {
            const int cn = cn0 + j * 16;
            if (cn < N) {
#pragma unroll
                for (int r = 0; r < 8; ++r) {
                    const int cm = cmBase + i * 16 + r;
                    float v = acc[i][j][r];
                    if constexpr (EPI == EPI_F32) {
                        ((float*)Cout)[(size_t)cm * ldc + cn] = v;
                    } else if constexpr (EPI == EPI_F16) {
                        ((_Float16*)Cout)[(size_t)cm * ldc + cn] = (_Float16)v;
                    } else {
                        ((_Float16*)Cout)[(size_t)cm * ldc + cn] =
                            (_Float16)(v > 0.f ? v : 0.f);
                    }
                }
            }
        }
    }
}

// ---------------------------------------------------------------------------
// Elementwise / helper kernels
// ---------------------------------------------------------------------------
__global__ __launch_bounds__(256)
void pack_f16(const float* __restrict__ src, _Float16* __restrict__ dst, size_t n)
{
    size_t i = (size_t)blockIdx.x * blockDim.x + threadIdx.x;
    if (i < n) dst[i] = (_Float16)src[i];
}

__global__ __launch_bounds__(256)
void pack_f16_2d(const float* __restrict__ src, _Float16* __restrict__ dst,
                 int rows, int cols, int src_ld)
{
    int i = blockIdx.x * blockDim.x + threadIdx.x;
    if (i < rows * cols) {
        int r = i / cols, c = i % cols;
        dst[(size_t)r * cols + c] = (_Float16)src[(size_t)r * src_ld + c];
    }
}

__global__ __launch_bounds__(256)
void negexp_kernel(const float* __restrict__ src, float* __restrict__ dst, int n)
{
    int i = blockIdx.x * blockDim.x + threadIdx.x;
    if (i < n) dst[i] = -__expf(src[i]);
}

// u = silu(depthwise_causal_conv(xi) + conv_b); xi = xz[:, 0:DI] token-major.
__global__ __launch_bounds__(256)
void conv_silu_kernel(const float* __restrict__ xz, const float* __restrict__ conv_w,
                      const float* __restrict__ conv_b, _Float16* __restrict__ u,
                      int Bn, int L, int DI)
{
    int idx = blockIdx.x * blockDim.x + threadIdx.x;
    if (idx >= Bn * L * DI) return;
    int d = idx % DI;
    int t = idx / DI;
    int l = t % L;
    int b = t / L;
    float acc = conv_b[d];
#pragma unroll
    for (int j = 0; j < 4; ++j) {
        int ll = l - 3 + j;
        if (ll >= 0)
            acc += xz[((size_t)(b * L + ll)) * (2 * DI) + d] * conv_w[d * 4 + j];
    }
    float s = acc / (1.f + __expf(-acc));
    u[(size_t)t * DI + d] = (_Float16)s;
}

// dusum[t] = sum_d Dp[d] * u[t,d]
__global__ __launch_bounds__(256)
void dusum_kernel(const _Float16* __restrict__ u, const float* __restrict__ Dp,
                  float* __restrict__ dusum, int DI)
{
    __shared__ float red[256];
    int t = blockIdx.x;
    float acc = 0.f;
    for (int d = threadIdx.x; d < DI; d += 256)
        acc += (float)u[(size_t)t * DI + d] * Dp[d];
    red[threadIdx.x] = acc;
    __syncthreads();
    for (int s = 128; s > 0; s >>= 1) {
        if (threadIdx.x < (unsigned)s) red[threadIdx.x] += red[threadIdx.x + s];
        __syncthreads();
    }
    if (threadIdx.x == 0) dusum[t] = red[0];
}

// Selective scan: 16 lanes per (b,d), one lane per state index n.
// Writes gated output into first half of y_cat (f16).
__global__ __launch_bounds__(256)
void scan_kernel(const float* __restrict__ dtf,    // (T, DI)
                 const _Float16* __restrict__ u,   // (T, DI)
                 const float* __restrict__ xdbl,   // (T, 96): [0:64)=dt_r [64:80)=B [80:96)=C
                 const float* __restrict__ xz,     // (T, 2*DI): z in [DI, 2*DI)
                 const float* __restrict__ Anorm,  // (DI, 16) = -exp(A_log)
                 const float* __restrict__ Dp, const float* __restrict__ b_dt,
                 _Float16* __restrict__ ycat,      // (T, 2*DI) first half
                 int Bn, int L, int DI)
{
    int gid = blockIdx.x * (blockDim.x >> 4) + (threadIdx.x >> 4);  // (b,d) group
    int n   = threadIdx.x & 15;
    if (gid >= Bn * DI) return;
    int b = gid / DI;
    int d = gid % DI;

    const float A_dn = Anorm[d * 16 + n];
    const float Dpd  = Dp[d];
    const float bdt  = b_dt[d];
    const int   DI2  = 2 * DI;
    float s = 0.f;

    for (int l = 0; l < L; ++l) {
        const int t = b * L + l;
        float dv    = dtf[(size_t)t * DI + d] + bdt;
        float delta = (dv > 20.f) ? dv : log1pf(__expf(dv));   // softplus
        float uv    = (float)u[(size_t)t * DI + d];
        float Bv    = xdbl[(size_t)t * 96 + 64 + n];
        float Cv    = xdbl[(size_t)t * 96 + 80 + n];
        s = __expf(delta * A_dn) * s + (delta * uv) * Bv;
        float y = s * Cv;
        y += __shfl_xor(y, 1);
        y += __shfl_xor(y, 2);
        y += __shfl_xor(y, 4);
        y += __shfl_xor(y, 8);
        if (n == 0) {
            float z  = xz[(size_t)t * DI2 + DI + d];
            float sz = z / (1.f + __expf(-z));
            ycat[(size_t)t * DI2 + d] = (_Float16)((y + Dpd * uv) * sz);
        }
    }
}

// y_t[t,d] = sum_n C[t,n]*A[d,n] + dusum[t] -> second half of y_cat (f16)
__global__ __launch_bounds__(256)
void yt_kernel(const float* __restrict__ xdbl, const float* __restrict__ Anorm,
               const float* __restrict__ dusum, _Float16* __restrict__ ycat,
               int T, int DI)
{
    int idx = blockIdx.x * blockDim.x + threadIdx.x;
    if (idx >= T * DI) return;
    int d = idx % DI;
    int t = idx / DI;
    const float* C = xdbl + (size_t)t * 96 + 80;
    float acc = dusum[t];
#pragma unroll
    for (int n = 0; n < 16; ++n) acc += C[n] * Anorm[d * 16 + n];
    ycat[(size_t)t * (2 * DI) + DI + d] = (_Float16)acc;
}

// ---------------------------------------------------------------------------
// Orchestration
// ---------------------------------------------------------------------------
extern "C" void kernel_launch(void* const* d_in, const int* in_sizes, int n_in,
                              void* d_out, int out_size, void* d_ws, size_t ws_size,
                              hipStream_t stream)
{
    (void)in_sizes; (void)n_in; (void)out_size; (void)ws_size;

    const float* x      = (const float*)d_in[0];
    const float* W_in   = (const float*)d_in[1];
    const float* conv_w = (const float*)d_in[2];
    const float* conv_b = (const float*)d_in[3];
    const float* W_x    = (const float*)d_in[4];
    const float* W_dt   = (const float*)d_in[5];
    const float* b_dt   = (const float*)d_in[6];
    const float* A_log  = (const float*)d_in[7];
    const float* Dp     = (const float*)d_in[8];
    const float* W_out1 = (const float*)d_in[9];
    const float* W_out2 = (const float*)d_in[10];
    float* out = (float*)d_out;

    const int Bn = 2, L = 2048, Dm = 1024, DI = 2048;
    const int T = Bn * L;          // 4096 tokens
    const int E2 = 2 * DI;         // 4096
    const int XD = 96, R = 64, S = 16;

    size_t off = 0;
    auto carve = [&](size_t bytes) -> void* {
        void* p = (char*)d_ws + off;
        off += (bytes + 255) & ~(size_t)255;
        return p;
    };
    _Float16* x_h   = (_Float16*)carve((size_t)T * Dm * 2);
    _Float16* Win_h = (_Float16*)carve((size_t)E2 * Dm * 2);
    _Float16* Wx_h  = (_Float16*)carve((size_t)XD * DI * 2);
    _Float16* Wdt_h = (_Float16*)carve((size_t)DI * R * 2);
    _Float16* Wo1_h = (_Float16*)carve((size_t)DI * E2 * 2);
    _Float16* Wo2_h = (_Float16*)carve((size_t)Dm * DI * 2);
    float*    xz    = (float*)carve((size_t)T * E2 * 4);
    _Float16* u_h   = (_Float16*)carve((size_t)T * DI * 2);
    float*    xdbl  = (float*)carve((size_t)T * XD * 4);
    _Float16* dtr_h = (_Float16*)carve((size_t)T * R * 2);
    float*    dtf   = (float*)carve((size_t)T * DI * 4);
    float*    Anorm = (float*)carve((size_t)DI * S * 4);
    float*    dusum = (float*)carve((size_t)T * 4);
    _Float16* ycat  = (_Float16*)carve((size_t)T * E2 * 2);
    _Float16* h_h   = (_Float16*)carve((size_t)T * DI * 2);

    const int NT = 256;
    auto cdiv = [](size_t a, size_t b) { return (unsigned)((a + b - 1) / b); };

    // ---- pack inputs/weights to f16 ----
    pack_f16<<<cdiv((size_t)T * Dm, NT), NT, 0, stream>>>(x, x_h, (size_t)T * Dm);
    pack_f16<<<cdiv((size_t)E2 * Dm, NT), NT, 0, stream>>>(W_in, Win_h, (size_t)E2 * Dm);
    pack_f16<<<cdiv((size_t)XD * DI, NT), NT, 0, stream>>>(W_x, Wx_h, (size_t)XD * DI);
    pack_f16<<<cdiv((size_t)DI * R, NT), NT, 0, stream>>>(W_dt, Wdt_h, (size_t)DI * R);
    pack_f16<<<cdiv((size_t)DI * E2, NT), NT, 0, stream>>>(W_out1, Wo1_h, (size_t)DI * E2);
    pack_f16<<<cdiv((size_t)Dm * DI, NT), NT, 0, stream>>>(W_out2, Wo2_h, (size_t)Dm * DI);

    // ---- GEMM1: xz[t,e] = x[t,:] . W_in[e,:]  (M=4096,N=4096,K=1024) ----
    {
        dim3 g(E2 / GTN, T / GTM);
        gemm_wmma_f16<EPI_F32><<<g, NT, 0, stream>>>(x_h, Dm, Win_h, Dm, xz, E2, T, E2, Dm);
    }

    // ---- depthwise causal conv + SiLU -> u ----
    conv_silu_kernel<<<cdiv((size_t)T * DI, NT), NT, 0, stream>>>(xz, conv_w, conv_b, u_h, Bn, L, DI);

    // ---- GEMM2: x_dbl[t,e] = u[t,:] . W_x[e,:]  (M=4096,N=96,K=2048) ----
    {
        dim3 g((XD + GTN - 1) / GTN, T / GTM);
        gemm_wmma_f16<EPI_F32><<<g, NT, 0, stream>>>(u_h, DI, Wx_h, DI, xdbl, XD, T, XD, DI);
    }

    // ---- pack dt_r slice to f16 ----
    pack_f16_2d<<<cdiv((size_t)T * R, NT), NT, 0, stream>>>(xdbl, dtr_h, T, R, XD);

    // ---- GEMM3: dt[t,d] = dt_r[t,:] . W_dt[d,:]  (M=4096,N=2048,K=64) ----
    {
        dim3 g(DI / GTN, T / GTM);
        gemm_wmma_f16<EPI_F32><<<g, NT, 0, stream>>>(dtr_h, R, Wdt_h, R, dtf, DI, T, DI, R);
    }

    // ---- A = -exp(A_log); dusum; selective scan; y_t ----
    negexp_kernel<<<cdiv((size_t)DI * S, NT), NT, 0, stream>>>(A_log, Anorm, DI * S);
    dusum_kernel<<<T, NT, 0, stream>>>(u_h, Dp, dusum, DI);
    scan_kernel<<<(Bn * DI) / 16, NT, 0, stream>>>(dtf, u_h, xdbl, xz, Anorm, Dp, b_dt,
                                                   ycat, Bn, L, DI);
    yt_kernel<<<cdiv((size_t)T * DI, NT), NT, 0, stream>>>(xdbl, Anorm, dusum, ycat, T, DI);

    // ---- GEMM4: h = relu(y_cat . W_out1^T)  (M=4096,N=2048,K=4096) ----
    {
        dim3 g(DI / GTN, T / GTM);
        gemm_wmma_f16<EPI_RELU_F16><<<g, NT, 0, stream>>>(ycat, E2, Wo1_h, E2, h_h, DI, T, DI, E2);
    }

    // ---- GEMM5: out = h . W_out2^T  (M=4096,N=1024,K=2048) -> f32 d_out ----
    {
        dim3 g(Dm / GTN, T / GTM);
        gemm_wmma_f16<EPI_F32><<<g, NT, 0, stream>>>(h_h, DI, Wo2_h, DI, out, Dm, T, Dm, DI);
    }
}
